// AttentionDAF_19353122635790
// MI455X (gfx1250) — matrix operand order repaired and checked by reading, not compiled, not run on hardware
//
#include <hip/hip_runtime.h>
#include <hip/hip_bf16.h>

#define DEVINL __device__ __forceinline__

typedef __attribute__((ext_vector_type(16))) __bf16 v16bf;
typedef __attribute__((ext_vector_type(8)))  __bf16 v8bf;
typedef __attribute__((ext_vector_type(8)))  float  v8f;
typedef __attribute__((ext_vector_type(4)))  float  f32x4;

static constexpr int BATCH = 16;
static constexpr int CLEN  = 512;
static constexpr int QLEN  = 512;
static constexpr int HD    = 768;
static constexpr int FH    = 4 * HD;   // 3072

// LDS tile strides (bf16 elements); padded so fragment b128 loads stay
// 16-byte aligned.
static constexpr int LDA = 40;   // A tile: 128 x 32, row stride 40
static constexpr int LDB = 136;  // B tile: 32 x 128 [k][n], row stride 136

DEVINL v8f vzero8() {
  v8f z;
#pragma unroll
  for (int i = 0; i < 8; ++i) z[i] = 0.f;
  return z;
}

DEVINL v8f wmma_bf16(v16bf a, v16bf b, v8f c) {
  return __builtin_amdgcn_wmma_f32_16x16x32_bf16(
      false, a, false, b, (short)0, c, false, false);
}

// ---- CDNA5 async global->LDS copy (ASYNCcnt-tracked, bypasses VGPRs) ------
DEVINL void async_copy_b128(unsigned lds_off, const void* gaddr) {
  asm volatile("global_load_async_to_lds_b128 %0, %1, off"
               :: "v"(lds_off), "v"(gaddr) : "memory");
}
DEVINL unsigned lds_off32(const void* p) {
  return (unsigned)(unsigned long long)p;   // low 32 bits = LDS offset
}

// A fragment, 16x32 bf16 per ISA layout.
DEVINL v16bf load_a_frag(const __bf16* As, int m0, int lane) {
  int m = m0 + (lane & 15);
  int h = lane >> 4;
  v8bf lo = *(const v8bf*)(As + m * LDA + h * 8);
  v8bf hi = *(const v8bf*)(As + m * LDA + 16 + h * 8);
  return __builtin_shufflevector(lo, hi, 0,1,2,3,4,5,6,7,8,9,10,11,12,13,14,15);
}

// B fragment, 32x16 bf16: lane%16 = N column, 16 elems = K in table order (Bs is [k][n])
DEVINL v16bf load_b_frag(const __bf16* Bs, int n0, int lane) {
  int n = n0 + (lane & 15);
  int h = lane >> 4;
  v16bf v;
#pragma unroll
  for (int s = 0; s < 8; ++s) {
    v[s]     = Bs[(h * 8 + s) * LDB + n];
    v[8 + s] = Bs[(16 + h * 8 + s) * LDB + n];
  }
  return v;
}

// ---- block reductions (wave32) -------------------------------------------
DEVINL float warp_sum(float v) {
#pragma unroll
  for (int o = 16; o > 0; o >>= 1) v += __shfl_down(v, o, 32);
  return v;
}
DEVINL float warp_max(float v) {
#pragma unroll
  for (int o = 16; o > 0; o >>= 1) v = fmaxf(v, __shfl_down(v, o, 32));
  return v;
}
DEVINL float block_sum(float v, float* sb) {
  int lane = threadIdx.x & 31, w = threadIdx.x >> 5;
  v = warp_sum(v);
  if (lane == 0) sb[w] = v;
  __syncthreads();
  if (threadIdx.x == 0) {
    float s = 0.f;
#pragma unroll
    for (int i = 0; i < 8; ++i) s += sb[i];
    sb[0] = s;
  }
  __syncthreads();
  float r = sb[0];
  __syncthreads();
  return r;
}
DEVINL float block_max(float v, float* sb) {
  int lane = threadIdx.x & 31, w = threadIdx.x >> 5;
  v = warp_max(v);
  if (lane == 0) sb[w] = v;
  __syncthreads();
  if (threadIdx.x == 0) {
    float s = sb[0];
#pragma unroll
    for (int i = 1; i < 8; ++i) s = fmaxf(s, sb[i]);
    sb[0] = s;
  }
  __syncthreads();
  float r = sb[0];
  __syncthreads();
  return r;
}

// ===========================================================================
// K1: prep — cw_bf16 = (c*wcq) as bf16, q_bf16 = q, s_c / s_q row dots.
// ===========================================================================
__global__ __launch_bounds__(256)
void prep_kernel(const float* __restrict__ inputs, const float* __restrict__ states,
                 const float* __restrict__ wc_w, const float* __restrict__ wc_b,
                 const float* __restrict__ wq_w, const float* __restrict__ wq_b,
                 const float* __restrict__ wcq_w, const float* __restrict__ wcq_b,
                 unsigned short* __restrict__ cw_us, unsigned short* __restrict__ qb_us,
                 float* __restrict__ s_c, float* __restrict__ s_q) {
  __shared__ float sb[8];
  __bf16* cw = (__bf16*)cw_us;
  __bf16* qb = (__bf16*)qb_us;
  int r = blockIdx.x, t = threadIdx.x;
  if (r < BATCH * CLEN) {
    const float* src = inputs + (size_t)r * HD;
    float partial = 0.f;
#pragma unroll
    for (int e = t; e < HD; e += 256) {
      float v = src[e];
      partial += v * wc_w[e];
      cw[(size_t)r * HD + e] = (__bf16)(v * wcq_w[e]);
    }
    float s = block_sum(partial, sb);
    if (t == 0) s_c[r] = s + wc_b[0] + wcq_b[0];
  } else {
    int rq = r - BATCH * CLEN;
    const float* src = states + (size_t)rq * HD;
    float partial = 0.f;
#pragma unroll
    for (int e = t; e < HD; e += 256) {
      float v = src[e];
      partial += v * wq_w[e];
      qb[(size_t)rq * HD + e] = (__bf16)v;
    }
    float s = block_sum(partial, sb);
    if (t == 0) s_q[rq] = s + wq_b[0];
  }
}

// ===========================================================================
// K2: gemm1 — S = cw_bf16 @ q_bf16^T + s_c + s_q + mask.
// Double-buffered: A via async DMA (issued a tile ahead, overlapped with
// WMMA), B staged in registers and transpose-scattered post-compute.
// ===========================================================================
__global__ __launch_bounds__(256)
void gemm1_scores(const unsigned short* __restrict__ cw_us,
                  const unsigned short* __restrict__ qb_us,
                  const float* __restrict__ s_c, const float* __restrict__ s_q,
                  const float* __restrict__ mask, float* __restrict__ S) {
  __shared__ __bf16 As[2][128 * LDA];
  __shared__ __bf16 Bs[2][32 * LDB];
  const __bf16* cw = (const __bf16*)cw_us;
  const __bf16* qb = (const __bf16*)qb_us;

  int b = blockIdx.z;
  int row0 = blockIdx.x * 128, col0 = blockIdx.y * 128;
  int t = threadIdx.x, lane = t & 31, wid = t >> 5;
  int wm = wid & 3, wn = wid >> 2;   // 4 x 2 waves, 32x64 each

  const __bf16* Ag = cw + ((size_t)b * CLEN + row0) * HD;
  const __bf16* Bg = qb + ((size_t)b * QLEN + col0) * HD;

  v8f acc[2][4];
#pragma unroll
  for (int mi = 0; mi < 2; ++mi)
#pragma unroll
    for (int ni = 0; ni < 4; ++ni) acc[mi][ni] = vzero8();

  auto loadB = [&](int kt, v8bf br[2]) {
#pragma unroll
    for (int li = 0; li < 2; ++li) {
      int idx = t + 256 * li;
      int n = idx >> 2, sg = idx & 3;
      br[li] = *(const v8bf*)(Bg + (size_t)n * HD + kt + sg * 8);
    }
  };
  auto asyncA = [&](int kt, int buf) {
#pragma unroll
    for (int li = 0; li < 2; ++li) {
      int idx = t + 256 * li;
      int r = idx >> 2, sg = idx & 3;
      async_copy_b128(lds_off32(&As[buf][r * LDA + sg * 8]),
                      Ag + (size_t)r * HD + kt + sg * 8);
    }
  };
  auto scatterB = [&](const v8bf br[2], int buf) {
#pragma unroll
    for (int li = 0; li < 2; ++li) {
      int idx = t + 256 * li;
      int n = idx >> 2, sg = idx & 3;
#pragma unroll
      for (int e = 0; e < 8; ++e) Bs[buf][(sg * 8 + e) * LDB + n] = br[li][e];
    }
  };
  auto compute = [&](int buf) {
    v16bf a0 = load_a_frag(&As[buf][0], wm * 32, lane);
    v16bf a1 = load_a_frag(&As[buf][0], wm * 32 + 16, lane);
#pragma unroll
    for (int ni = 0; ni < 4; ++ni) {
      v16bf bf = load_b_frag(&Bs[buf][0], wn * 64 + ni * 16, lane);
      acc[0][ni] = wmma_bf16(a0, bf, acc[0][ni]);
      acc[1][ni] = wmma_bf16(a1, bf, acc[1][ni]);
    }
  };

  const int NT = HD / 32;   // 24
  {
    v8bf br0[2];
    loadB(0, br0);
    asyncA(0, 0);
    scatterB(br0, 0);
  }
#pragma unroll 2
  for (int it = 0; it < NT - 1; ++it) {
    int cur = it & 1, nxt = cur ^ 1;
    v8bf brn[2];
    loadB((it + 1) * 32, brn);
    asyncA((it + 1) * 32, nxt);
    asm volatile("s_wait_asynccnt 0x2" ::: "memory");  // tile `cur` resident
    __syncthreads();
    compute(cur);
    scatterB(brn, nxt);
    __syncthreads();
  }
  asm volatile("s_wait_asynccnt 0x0" ::: "memory");
  __syncthreads();
  compute((NT - 1) & 1);

  int hi = lane >> 4, col = lane & 15;
#pragma unroll
  for (int mi = 0; mi < 2; ++mi) {
    int r0 = row0 + wm * 32 + mi * 16 + hi * 8;
    float sc[8];
#pragma unroll
    for (int e = 0; e < 8; ++e) sc[e] = s_c[b * CLEN + r0 + e];
#pragma unroll
    for (int ni = 0; ni < 4; ++ni) {
      int c = col0 + wn * 64 + ni * 16 + col;
      float sq = s_q[b * QLEN + c];
#pragma unroll
      for (int e = 0; e < 8; ++e) {
        size_t off = ((size_t)b * CLEN + r0 + e) * QLEN + c;
        S[off] = acc[mi][ni][e] + sc[e] + sq + mask[off];
      }
    }
  }
}

// ===========================================================================
// K3: softmax over j per (b,i) row; saves raw row max; writes A as bf16
// ===========================================================================
__global__ __launch_bounds__(256)
void softmax_kernel(const float* __restrict__ S, float* __restrict__ m_buf,
                    unsigned short* __restrict__ A_us) {
  __shared__ float sb[8];
  __bf16* A = (__bf16*)A_us;
  int r = blockIdx.x, t = threadIdx.x;
  const float* row = S + (size_t)r * QLEN;
  float x0 = row[t], x1 = row[t + 256];
  float mx = block_max(fmaxf(x0, x1), sb);
  if (t == 0) m_buf[r] = mx;
  float e0 = __expf(x0 - mx), e1 = __expf(x1 - mx);
  float sum = block_sum(e0 + e1, sb);
  float inv = 1.f / sum;
  A[(size_t)r * QLEN + t]       = (__bf16)(e0 * inv);
  A[(size_t)r * QLEN + t + 256] = (__bf16)(e1 * inv);
}

// ===========================================================================
// K4: b_att = softmax_i(m[b,:]); q2c[b,h] = sum_i b_att[i] * c[b,i,h]
// ===========================================================================
__global__ __launch_bounds__(256)
void batt_kernel(const float* __restrict__ m_buf, const float* __restrict__ inputs,
                 float* __restrict__ q2c) {
  __shared__ float sb[8];
  __shared__ float batt[CLEN];
  int b = blockIdx.x, t = threadIdx.x;
  float m0 = m_buf[b * CLEN + t], m1 = m_buf[b * CLEN + t + 256];
  float mx = block_max(fmaxf(m0, m1), sb);
  float e0 = __expf(m0 - mx), e1 = __expf(m1 - mx);
  float sum = block_sum(e0 + e1, sb);
  float inv = 1.f / sum;
  batt[t] = e0 * inv;
  batt[t + 256] = e1 * inv;
  __syncthreads();
  for (int h = t; h < HD; h += 256) {
    const float* cb = inputs + ((size_t)b * CLEN) * HD + h;
    float acc = 0.f;
#pragma unroll 4
    for (int i = 0; i < CLEN; ++i) acc += batt[i] * cb[(size_t)i * HD];
    q2c[b * HD + h] = acc;
  }
}

// ===========================================================================
// K5: gemm2 — c2q = A_bf16 @ q_bf16, bf16 result.
// Fully async double buffer: 4 DMA ops per wave per tile.
// ===========================================================================
__global__ __launch_bounds__(256)
void gemm2_c2q(const unsigned short* __restrict__ A_us,
               const unsigned short* __restrict__ qb_us,
               unsigned short* __restrict__ c2q_us) {
  __shared__ __bf16 As[2][128 * LDA];
  __shared__ __bf16 Bs[2][32 * LDB];
  const __bf16* Ab = (const __bf16*)A_us;
  const __bf16* qb = (const __bf16*)qb_us;
  __bf16* c2q = (__bf16*)c2q_us;

  int b = blockIdx.z;
  int row0 = blockIdx.x * 128, col0 = blockIdx.y * 128;
  int t = threadIdx.x, lane = t & 31, wid = t >> 5;
  int wm = wid & 3, wn = wid >> 2;

  const __bf16* Ag = Ab + ((size_t)b * CLEN + row0) * QLEN;
  const __bf16* Bg = qb + (size_t)b * QLEN * HD;

  v8f acc[2][4];
#pragma unroll
  for (int mi = 0; mi < 2; ++mi)
#pragma unroll
    for (int ni = 0; ni < 4; ++ni) acc[mi][ni] = vzero8();

  auto asyncTile = [&](int kt, int buf) {
#pragma unroll
    for (int li = 0; li < 2; ++li) {
      int idx = t + 256 * li;
      int r = idx >> 2, sg = idx & 3;
      async_copy_b128(lds_off32(&As[buf][r * LDA + sg * 8]),
                      Ag + (size_t)r * QLEN + kt + sg * 8);
    }
#pragma unroll
    for (int li = 0; li < 2; ++li) {
      int idx = t + 256 * li;
      int k = idx >> 4, sg = idx & 15;
      async_copy_b128(lds_off32(&Bs[buf][k * LDB + sg * 8]),
                      Bg + (size_t)(kt + k) * HD + col0 + sg * 8);
    }
  };
  auto compute = [&](int buf) {
    v16bf a0 = load_a_frag(&As[buf][0], wm * 32, lane);
    v16bf a1 = load_a_frag(&As[buf][0], wm * 32 + 16, lane);
#pragma unroll
    for (int ni = 0; ni < 4; ++ni) {
      v16bf bf = load_b_frag(&Bs[buf][0], wn * 64 + ni * 16, lane);
      acc[0][ni] = wmma_bf16(a0, bf, acc[0][ni]);
      acc[1][ni] = wmma_bf16(a1, bf, acc[1][ni]);
    }
  };

  const int NT = QLEN / 32;   // 16
  asyncTile(0, 0);
#pragma unroll 2
  for (int it = 0; it < NT - 1; ++it) {
    int cur = it & 1, nxt = cur ^ 1;
    asyncTile((it + 1) * 32, nxt);
    asm volatile("s_wait_asynccnt 0x4" ::: "memory");  // tile `cur` resident
    __syncthreads();
    compute(cur);
    __syncthreads();
  }
  asm volatile("s_wait_asynccnt 0x0" ::: "memory");
  __syncthreads();
  compute((NT - 1) & 1);

  int hi = lane >> 4, col = lane & 15;
#pragma unroll
  for (int mi = 0; mi < 2; ++mi)
#pragma unroll
    for (int ni = 0; ni < 4; ++ni) {
      int r0 = row0 + wm * 32 + mi * 16 + hi * 8;
      int c  = col0 + wn * 64 + ni * 16 + col;
#pragma unroll
      for (int e = 0; e < 8; ++e) {
        int r = r0 + e;
        c2q[((size_t)b * CLEN + r) * HD + c] = (__bf16)acc[mi][ni][e];
      }
    }
}

// ===========================================================================
// K6: gemm3 — y = relu(X @ wa^T + wa_b) + c,  X generated on the fly.
// Register-staged double buffer, single barrier per K chunk:
//   stage loads (overlap compute) -> barrier -> WMMA(cur) -> commit(nxt)
// ===========================================================================
__global__ __launch_bounds__(256)
void gemm3_out(const float* __restrict__ c_in, const unsigned short* __restrict__ c2q_us,
               const float* __restrict__ q2c, const float* __restrict__ wa_w,
               const float* __restrict__ wa_b, float* __restrict__ y) {
  __shared__ __bf16 As[2][128 * LDA];
  __shared__ __bf16 Bs[2][32 * LDB];
  const __bf16* c2q = (const __bf16*)c2q_us;

  int row0 = blockIdx.x * 128;   // 512 % 128 == 0 -> single batch per tile
  int col0 = blockIdx.y * 128;
  int b = row0 >> 9;
  int t = threadIdx.x, lane = t & 31, wid = t >> 5;
  int wm = wid & 3, wn = wid >> 2;

  v8f acc[2][4];
#pragma unroll
  for (int mi = 0; mi < 2; ++mi)
#pragma unroll
    for (int ni = 0; ni < 4; ++ni) acc[mi][ni] = vzero8();

  auto stage = [&](int kt, v8bf ar[2], v8bf br[2]) {
    int region = kt / HD;         // uniform per chunk (32 | 768)
    int f0 = kt - region * HD;
#pragma unroll
    for (int li = 0; li < 2; ++li) {
      int idx = t + 256 * li;
      int r = idx >> 2, sg = idx & 3;
      int gr = row0 + r;
      int fo = f0 + sg * 8;
      v8bf o;
      if (region == 1) {
        o = *(const v8bf*)(c2q + (size_t)gr * HD + fo);
      } else {
        f32x4 c0 = *(const f32x4*)(c_in + (size_t)gr * HD + fo);
        f32x4 c1 = *(const f32x4*)(c_in + (size_t)gr * HD + fo + 4);
        if (region == 0) {
#pragma unroll
          for (int e = 0; e < 4; ++e) { o[e] = (__bf16)c0[e]; o[4 + e] = (__bf16)c1[e]; }
        } else if (region == 2) {
          v8bf cq = *(const v8bf*)(c2q + (size_t)gr * HD + fo);
#pragma unroll
          for (int e = 0; e < 4; ++e) {
            o[e]     = (__bf16)(c0[e] * (float)cq[e]);
            o[4 + e] = (__bf16)(c1[e] * (float)cq[4 + e]);
          }
        } else {
          f32x4 g0 = *(const f32x4*)(q2c + b * HD + fo);
          f32x4 g1 = *(const f32x4*)(q2c + b * HD + fo + 4);
#pragma unroll
          for (int e = 0; e < 4; ++e) {
            o[e]     = (__bf16)(c0[e] * g0[e]);
            o[4 + e] = (__bf16)(c1[e] * g1[e]);
          }
        }
      }
      ar[li] = o;
    }
#pragma unroll
    for (int li = 0; li < 2; ++li) {
      int idx = t + 256 * li;
      int hl = idx >> 2, sg = idx & 3;
      int h = col0 + hl;
      const float* wp = wa_w + (size_t)h * FH + kt + sg * 8;
      f32x4 w0 = *(const f32x4*)wp;
      f32x4 w1 = *(const f32x4*)(wp + 4);
      v8bf o;
#pragma unroll
      for (int e = 0; e < 4; ++e) { o[e] = (__bf16)w0[e]; o[4 + e] = (__bf16)w1[e]; }
      br[li] = o;
    }
  };
  auto commit = [&](const v8bf ar[2], const v8bf br[2], int buf) {
#pragma unroll
    for (int li = 0; li < 2; ++li) {
      int idx = t + 256 * li;
      int r = idx >> 2, sg = idx & 3;
      *(v8bf*)&As[buf][r * LDA + sg * 8] = ar[li];
    }
#pragma unroll
    for (int li = 0; li < 2; ++li) {
      int idx = t + 256 * li;
      int hl = idx >> 2, sg = idx & 3;
#pragma unroll
      for (int e = 0; e < 8; ++e) Bs[buf][(sg * 8 + e) * LDB + hl] = br[li][e];
    }
  };
  auto compute = [&](int buf) {
    v16bf a0 = load_a_frag(&As[buf][0], wm * 32, lane);
    v16bf a1 = load_a_frag(&As[buf][0], wm * 32 + 16, lane);
#pragma unroll
    for (int ni = 0; ni < 4; ++ni) {
      v16bf bf = load_b_frag(&Bs[buf][0], wn * 64 + ni * 16, lane);
      acc[0][ni] = wmma_bf16(a0, bf, acc[0][ni]);
      acc[1][ni] = wmma_bf16(a1, bf, acc[1][ni]);
    }
  };

  const int NT = FH / 32;   // 96
  {
    v8bf a0r[2], b0r[2];
    stage(0, a0r, b0r);
    commit(a0r, b0r, 0);
  }
#pragma unroll 2
  for (int it = 0; it < NT - 1; ++it) {
    int cur = it & 1, nxt = cur ^ 1;
    v8bf ar[2], br[2];
    stage((it + 1) * 32, ar, br);
    __syncthreads();          // publish tile `cur`, fence buffer reuse
    compute(cur);
    commit(ar, br, nxt);
  }
  __syncthreads();
  compute((NT - 1) & 1);

  int hi = lane >> 4, col = lane & 15;
#pragma unroll
  for (int mi = 0; mi < 2; ++mi)
#pragma unroll
    for (int ni = 0; ni < 4; ++ni) {
      int r0 = row0 + wm * 32 + mi * 16 + hi * 8;
      int c  = col0 + wn * 64 + ni * 16 + col;
      float bias = wa_b[c];
#pragma unroll
      for (int e = 0; e < 8; ++e) {
        int r = r0 + e;
        float v = fmaxf(acc[mi][ni][e] + bias, 0.f) + c_in[(size_t)r * HD + c];
        y[(size_t)r * HD + c] = v;
      }
    }
}

// ===========================================================================
// K7: LayerNorm per row of y -> out
// ===========================================================================
__global__ __launch_bounds__(256)
void ln_kernel(const float* __restrict__ y, const float* __restrict__ g,
               const float* __restrict__ beta, float* __restrict__ out) {
  __shared__ float sb[8];
  int r = blockIdx.x, t = threadIdx.x;
  const float* yr = y + (size_t)r * HD;
  float v0 = yr[t], v1 = yr[t + 256], v2 = yr[t + 512];
  float mu = block_sum(v0 + v1 + v2, sb) * (1.f / HD);
  float d0 = v0 - mu, d1 = v1 - mu, d2 = v2 - mu;
  float var = block_sum(d0 * d0 + d1 * d1 + d2 * d2, sb) * (1.f / HD);
  float rs = rsqrtf(var + 1e-5f);
  float* orow = out + (size_t)r * HD;
  orow[t]       = d0 * rs * g[t]       + beta[t];
  orow[t + 256] = d1 * rs * g[t + 256] + beta[t + 256];
  orow[t + 512] = d2 * rs * g[t + 512] + beta[t + 512];
}

// ===========================================================================
extern "C" void kernel_launch(void* const* d_in, const int* in_sizes, int n_in,
                              void* d_out, int out_size, void* d_ws, size_t ws_size,
                              hipStream_t stream) {
  const float* inputs = (const float*)d_in[0];
  const float* states = (const float*)d_in[1];
  const float* mask   = (const float*)d_in[2];
  const float* wc_w   = (const float*)d_in[3];
  const float* wc_b   = (const float*)d_in[4];
  const float* wq_w   = (const float*)d_in[5];
  const float* wq_b   = (const float*)d_in[6];
  const float* wcq_w  = (const float*)d_in[7];
  const float* wcq_b  = (const float*)d_in[8];
  const float* wa_w   = (const float*)d_in[9];
  const float* wa_b   = (const float*)d_in[10];
  const float* ln_g   = (const float*)d_in[11];
  const float* ln_b   = (const float*)d_in[12];
  float* out = (float*)d_out;

  char* w = (char*)d_ws;
  size_t o = 0;
  unsigned short* cw_bf = (unsigned short*)(w + o); o += (size_t)BATCH * CLEN * HD * 2;
  unsigned short* q_bf  = (unsigned short*)(w + o); o += (size_t)BATCH * QLEN * HD * 2;
  float* s_c            = (float*)(w + o);          o += (size_t)BATCH * CLEN * 4;
  float* s_q            = (float*)(w + o);          o += (size_t)BATCH * QLEN * 4;
  float* Sbuf           = (float*)(w + o);          o += (size_t)BATCH * CLEN * QLEN * 4;
  float* m_buf          = (float*)(w + o);          o += (size_t)BATCH * CLEN * 4;
  unsigned short* A_bf  = (unsigned short*)(w + o); o += (size_t)BATCH * CLEN * QLEN * 2;
  float* q2c            = (float*)(w + o);          o += (size_t)BATCH * HD * 4;
  unsigned short* c2q   = (unsigned short*)(w + o); o += (size_t)BATCH * CLEN * HD * 2;
  float* ybuf           = (float*)(w + o);          o += (size_t)BATCH * CLEN * HD * 4;

  prep_kernel<<<dim3(BATCH * CLEN + BATCH * QLEN), 256, 0, stream>>>(
      inputs, states, wc_w, wc_b, wq_w, wq_b, wcq_w, wcq_b, cw_bf, q_bf, s_c, s_q);

  gemm1_scores<<<dim3(CLEN / 128, QLEN / 128, BATCH), 256, 0, stream>>>(
      cw_bf, q_bf, s_c, s_q, mask, Sbuf);

  softmax_kernel<<<dim3(BATCH * CLEN), 256, 0, stream>>>(Sbuf, m_buf, A_bf);

  batt_kernel<<<dim3(BATCH), 256, 0, stream>>>(m_buf, inputs, q2c);

  gemm2_c2q<<<dim3(CLEN / 128, HD / 128, BATCH), 256, 0, stream>>>(A_bf, q_bf, c2q);

  gemm3_out<<<dim3(BATCH * CLEN / 128, HD / 128), 256, 0, stream>>>(
      inputs, c2q, q2c, wa_w, wa_b, ybuf);

  ln_kernel<<<dim3(BATCH * CLEN), 256, 0, stream>>>(ybuf, ln_g, ln_b, out);
}
